// MSA1_7421703487726
// MI455X (gfx1250) — matrix operand order repaired and checked by reading, compiled, not run on hardware
//
#include <hip/hip_runtime.h>
#include <hip/hip_bf16.h>
#include <stdint.h>

typedef __attribute__((ext_vector_type(16))) _Float16 v16h;
typedef __attribute__((ext_vector_type(8)))  _Float16 v8h;
typedef __attribute__((ext_vector_type(8)))  float    v8f;

#define NHID     768
#define NHEAD    12
#define HEAD_DIM 64
#define WIN      5
#define BATCH    4
#define SEQ      2048
#define ALEN     5

#define MTILE    128
#define NTILE    64
#define BSTRIDE  40   // padded row (halves) of transposed B tile: 80B, 16B-aligned

// ---------------------------------------------------------------------------
// f32 -> f16 elementwise convert (weights only).
// ---------------------------------------------------------------------------
__global__ __launch_bounds__(256)
void cvt_f32_f16(const float* __restrict__ src, _Float16* __restrict__ dst, int n) {
    int i = blockIdx.x * 256 + threadIdx.x;
    if (i < n) dst[i] = (_Float16)src[i];
}

// ---------------------------------------------------------------------------
// WMMA GEMM:  C[b] = A(768x768 f16 row-major) * B[b](768 x N row-major) + bias
// Block: 256 threads = 8 wave32s; C tile 128x64; each wave: 2 A-frags x
// 2 B-frags -> 4x v_wmma_f32_16x16x32_f16 per 32-wide K-step.
// A tile (128x32 halves) DMA'd global->LDS via GLOBAL_LOAD_ASYNC_TO_LDS_B128,
// triple-buffered, next tile left in flight during compute (ASYNCcnt).
// B tile converted f32->f16 through registers, double-buffered.
// One barrier per K-step; barrier waits only DScnt+ASYNCcnt (not LOADcnt), so
// next-tile B global loads stay in flight across the barrier.
// ---------------------------------------------------------------------------
template <typename BT, typename OT>
__global__ __launch_bounds__(256)
void wmma_gemm(const _Float16* __restrict__ A,
               const BT*       __restrict__ Bsrc,
               const float*    __restrict__ bias,
               OT*             __restrict__ C,
               int N, long strideB, long strideC)
{
    const int K     = NHID;                // 768
    const int NSTEP = K / 32;              // 24
    const int tid   = threadIdx.x;
    const int lane  = tid & 31;
    const int w     = tid >> 5;            // wave id 0..7
    const int m0    = blockIdx.y * MTILE;
    const int n0    = blockIdx.x * NTILE;

    Bsrc += (long)blockIdx.z * strideB;
    C    += (long)blockIdx.z * strideC;

    __shared__ __align__(16) _Float16 lsA[3 * MTILE * 32];     // [buf][m][k]
    __shared__ __align__(16) _Float16 lsB[2 * NTILE * BSTRIDE];// [buf][n][k] (transposed)

    const int subM = (w & 3) * 32;         // wave covers rows subM..subM+31
    const int subN = (w >> 2) * 32;        // wave covers cols subN..subN+31

    v8f acc00 = {}, acc01 = {}, acc10 = {}, acc11 = {};

    // cooperative A loader: two 16B async DMAs per thread (128x32 halves = 8KB)
    const int arow = tid >> 2;             // 0..63 (second chunk: +64)
    const int acol = (tid & 3) * 8;        // 0,8,16,24
    // cooperative B loader: 8 elements per thread (convert + transpose)
    const int bk   = tid & 31;
    const int bn   = (tid >> 5) * 8;

    const int r    = lane & 15;
    const int akb  = (lane < 16) ? 0 : 8;  // A frag K base (ISA 16-bit A layout)
    const int bkb  = (lane < 16) ? 0 : 8;  // B frag K base (same table order as A)

    // --- helpers -----------------------------------------------------------
    auto issueA = [&](int kk, _Float16* buf) {
        uint32_t l0 = (uint32_t)(uintptr_t)(buf + arow * 32 + acol);
        uint64_t g0 = (uint64_t)(uintptr_t)(A + (long)(m0 + arow) * K + kk + acol);
        asm volatile("global_load_async_to_lds_b128 %0, %1, off"
                     :: "v"(l0), "v"(g0) : "memory");
        uint32_t l1 = (uint32_t)(uintptr_t)(buf + (arow + 64) * 32 + acol);
        uint64_t g1 = (uint64_t)(uintptr_t)(A + (long)(m0 + arow + 64) * K + kk + acol);
        asm volatile("global_load_async_to_lds_b128 %0, %1, off"
                     :: "v"(l1), "v"(g1) : "memory");
    };
    auto loadB = [&](int kk, float* dst) {
        #pragma unroll
        for (int i = 0; i < 8; ++i)
            dst[i] = (float)Bsrc[(long)(kk + bk) * N + n0 + bn + i];
    };
    auto ldFrag = [&](const _Float16* p) -> v16h {
        v8h lo = *(const v8h*)p;
        v8h hi = *(const v8h*)(p + 16);
        return __builtin_shufflevector(lo, hi,
                 0,1,2,3,4,5,6,7,8,9,10,11,12,13,14,15);
    };
    auto ldFragB = [&](const _Float16* p) -> v16h {
        v8h lo = *(const v8h*)p;
        v8h hi = *(const v8h*)(p + 16);
        return __builtin_shufflevector(lo, hi,
                 0,1,2,3,4,5,6,7,8,9,10,11,12,13,14,15);
    };

    // --- prologue ----------------------------------------------------------
    issueA(0, lsA);
    float bt[8];
    loadB(0, bt);

    // --- main K pipeline ---------------------------------------------------
    for (int s = 0; s < NSTEP; ++s) {
        _Float16* Acur = lsA + (s % 3) * (MTILE * 32);
        _Float16* Bcur = lsB + (s & 1) * (NTILE * BSTRIDE);

        if (s + 1 < NSTEP)                       // next A tile: DMA in flight
            issueA((s + 1) * 32, lsA + ((s + 1) % 3) * (MTILE * 32));

        #pragma unroll
        for (int i = 0; i < 8; ++i)              // commit current B tile
            Bcur[(bn + i) * BSTRIDE + bk] = (_Float16)bt[i];

        if (s + 1 < NSTEP)                       // next B tile into registers
            loadB((s + 1) * 32, bt);
        if (s + 2 < NSTEP)
            __builtin_prefetch(Bsrc + (long)((s + 2) * 32 + bk) * N + n0 + bn, 0, 1);

        // current A tile resident (next one may stay in flight: 2 DMAs)
        if (s + 1 < NSTEP) asm volatile("s_wait_asynccnt 0x2" ::: "memory");
        else               asm volatile("s_wait_asynccnt 0x0" ::: "memory");
        // barrier draining only DScnt (LDS stores); B global loads stay in flight
        asm volatile("s_wait_dscnt 0x0\n\t"
                     "s_barrier_signal -1\n\t"
                     "s_barrier_wait -1" ::: "memory");

        // fragments
        v16h a0  = ldFrag (Acur + (subM + r)      * 32 + akb);
        v16h a1  = ldFrag (Acur + (subM + 16 + r) * 32 + akb);
        v16h b0  = ldFragB(Bcur + (subN + r)      * BSTRIDE + bkb);
        v16h b1  = ldFragB(Bcur + (subN + 16 + r) * BSTRIDE + bkb);

        acc00 = __builtin_amdgcn_wmma_f32_16x16x32_f16(
                    false, a0, false, b0, (short)0, acc00, false, false);
        acc01 = __builtin_amdgcn_wmma_f32_16x16x32_f16(
                    false, a0, false, b1, (short)0, acc01, false, false);
        acc10 = __builtin_amdgcn_wmma_f32_16x16x32_f16(
                    false, a1, false, b0, (short)0, acc10, false, false);
        acc11 = __builtin_amdgcn_wmma_f32_16x16x32_f16(
                    false, a1, false, b1, (short)0, acc11, false, false);
    }

    // C layout: VGPR i -> M = i (lanes 0-15) or i+8 (lanes 16-31), N = lane%16
    const int mb = (lane < 16) ? 0 : 8;
    const int cn = n0 + subN + r;
    #pragma unroll
    for (int i = 0; i < 8; ++i) {
        int ma = m0 + subM + mb + i;        // rows of A-frag 0
        int mbig = ma + 16;                 // rows of A-frag 1
        float bva = bias[ma];
        float bvb = bias[mbig];
        C[(long)ma   * N + cn]      = (OT)(acc00[i] + bva);
        C[(long)ma   * N + cn + 16] = (OT)(acc01[i] + bva);
        C[(long)mbig * N + cn]      = (OT)(acc10[i] + bvb);
        C[(long)mbig * N + cn + 16] = (OT)(acc11[i] + bvb);
    }
}

// ---------------------------------------------------------------------------
// Attention: one thread per (b, h, l). Zero-padded windows contribute score
// exactly 0 (matching reference zero-pad semantics). q cached in registers;
// all accesses coalesced along l.
// ---------------------------------------------------------------------------
__global__ __launch_bounds__(256)
void attn_kernel(const _Float16* __restrict__ q, const _Float16* __restrict__ k,
                 const _Float16* __restrict__ v, const _Float16* __restrict__ ak,
                 const _Float16* __restrict__ av, _Float16* __restrict__ att)
{
    const int l = blockIdx.x * 256 + threadIdx.x;
    const int h = blockIdx.y;
    const int b = blockIdx.z;

    const long base  = ((long)b * NHID + (long)h * HEAD_DIM) * SEQ;
    const long abase = ((long)b * NHID + (long)h * HEAD_DIM) * (ALEN * SEQ);
    const long achs  = (long)ALEN * SEQ;

    float qr[HEAD_DIM];
    #pragma unroll
    for (int c = 0; c < HEAD_DIM; ++c)
        qr[c] = (float)q[base + (long)c * SEQ + l];

    float sc[WIN + ALEN];
    float msk[WIN];
    int   cps[WIN];
    #pragma unroll
    for (int wj = 0; wj < WIN; ++wj) {
        int pos  = l + wj - WIN / 2;
        bool ok  = (pos >= 0) && (pos < SEQ);
        msk[wj]  = ok ? 1.0f : 0.0f;
        cps[wj]  = ok ? pos : l;
        float s  = 0.0f;
        #pragma unroll
        for (int c = 0; c < HEAD_DIM; ++c)
            s += qr[c] * (float)k[base + (long)c * SEQ + cps[wj]];
        sc[wj] = ok ? s * 0.125f : 0.0f;
    }
    #pragma unroll
    for (int a = 0; a < ALEN; ++a) {
        float s = 0.0f;
        #pragma unroll
        for (int c = 0; c < HEAD_DIM; ++c)
            s += qr[c] * (float)ak[abase + (long)c * achs + (long)a * SEQ + l];
        sc[WIN + a] = s * 0.125f;
    }

    float mx = sc[0];
    #pragma unroll
    for (int j = 1; j < WIN + ALEN; ++j) mx = fmaxf(mx, sc[j]);
    float al[WIN + ALEN];
    float den = 0.0f;
    #pragma unroll
    for (int j = 0; j < WIN + ALEN; ++j) { al[j] = __expf(sc[j] - mx); den += al[j]; }
    float inv = 1.0f / den;
    #pragma unroll
    for (int j = 0; j < WIN + ALEN; ++j) al[j] *= inv;

    #pragma unroll 4
    for (int c = 0; c < HEAD_DIM; ++c) {
        float acc = 0.0f;
        #pragma unroll
        for (int wj = 0; wj < WIN; ++wj)
            acc += al[wj] * msk[wj] * (float)v[base + (long)c * SEQ + cps[wj]];
        #pragma unroll
        for (int a = 0; a < ALEN; ++a)
            acc += al[WIN + a] * (float)av[abase + (long)c * achs + (long)a * SEQ + l];
        att[base + (long)c * SEQ + l] = (_Float16)acc;
    }
}

// ---------------------------------------------------------------------------
extern "C" void kernel_launch(void* const* d_in, const int* in_sizes, int n_in,
                              void* d_out, int out_size, void* d_ws, size_t ws_size,
                              hipStream_t stream) {
    const float* x  = (const float*)d_in[0];
    const float* ax = (const float*)d_in[1];
    const float* Wq = (const float*)d_in[2];
    const float* bq = (const float*)d_in[3];
    const float* Wk = (const float*)d_in[4];
    const float* bk = (const float*)d_in[5];
    const float* Wv = (const float*)d_in[6];
    const float* bv = (const float*)d_in[7];
    const float* Wo = (const float*)d_in[8];
    const float* bo = (const float*)d_in[9];
    float* out = (float*)d_out;

    char* ws = (char*)d_ws;
    size_t off = 0;
    auto alloc16 = [&](size_t halves) {
        _Float16* p = (_Float16*)(ws + off);
        off += halves * sizeof(_Float16);
        return p;
    };
    const size_t WSZ  = (size_t)NHID * NHID;
    const size_t XSZ  = (size_t)BATCH * NHID * SEQ;
    const size_t AXSZ = (size_t)BATCH * NHID * ALEN * SEQ;

    _Float16* Wq16  = alloc16(WSZ);
    _Float16* Wk16  = alloc16(WSZ);
    _Float16* Wv16  = alloc16(WSZ);
    _Float16* Wo16  = alloc16(WSZ);
    _Float16* q16   = alloc16(XSZ);
    _Float16* k16   = alloc16(XSZ);
    _Float16* v16   = alloc16(XSZ);
    _Float16* ak16  = alloc16(AXSZ);
    _Float16* av16  = alloc16(AXSZ);
    _Float16* att16 = alloc16(XSZ);
    (void)ws_size;

    {
        int n = (int)WSZ, g = (n + 255) / 256;
        cvt_f32_f16<<<g, 256, 0, stream>>>(Wq, Wq16, n);
        cvt_f32_f16<<<g, 256, 0, stream>>>(Wk, Wk16, n);
        cvt_f32_f16<<<g, 256, 0, stream>>>(Wv, Wv16, n);
        cvt_f32_f16<<<g, 256, 0, stream>>>(Wo, Wo16, n);
    }

    dim3 blk(256);
    const long sX  = (long)NHID * SEQ;
    const long sAX = (long)NHID * ALEN * SEQ;

    dim3 gq(SEQ / NTILE, NHID / MTILE, BATCH);
    wmma_gemm<float, _Float16><<<gq, blk, 0, stream>>>(Wq16, x, bq, q16, SEQ, sX, sX);
    wmma_gemm<float, _Float16><<<gq, blk, 0, stream>>>(Wk16, x, bk, k16, SEQ, sX, sX);
    wmma_gemm<float, _Float16><<<gq, blk, 0, stream>>>(Wv16, x, bv, v16, SEQ, sX, sX);

    dim3 ga(ALEN * SEQ / NTILE, NHID / MTILE, BATCH);
    wmma_gemm<float, _Float16><<<ga, blk, 0, stream>>>(Wk16, ax, bk, ak16, ALEN * SEQ, sAX, sAX);
    wmma_gemm<float, _Float16><<<ga, blk, 0, stream>>>(Wv16, ax, bv, av16, ALEN * SEQ, sAX, sAX);

    dim3 gat(SEQ / 256, NHEAD, BATCH);
    attn_kernel<<<gat, blk, 0, stream>>>(q16, k16, v16, ak16, av16, att16);

    wmma_gemm<_Float16, float><<<gq, blk, 0, stream>>>(Wo16, att16, bo, out, SEQ, sX, sX);
}
